// IARM_89489938579708
// MI455X (gfx1250) — matrix operand order repaired and checked by reading, not compiled, not run on hardware
//
#include <hip/hip_runtime.h>

typedef __attribute__((ext_vector_type(16))) _Float16 v16h;
typedef __attribute__((ext_vector_type(8)))  _Float16 v8h;
typedef __attribute__((ext_vector_type(8)))  float    v8f;
typedef __attribute__((ext_vector_type(2)))  float    f2v;

#define ALPHA_SLOPE 0.2f
#define NEGV -1e9f

constexpr int B = 32, N = 512, D = 768, R = 128, H = 4, NLAYER = 2;
constexpr int HR = H * R;          // 512
constexpr size_t ATTN_LDS = (size_t)512 * R * 2     // Wh tile, fragment-packed
                          + (size_t)4 * 16 * 512 * 2 // per-wave p tiles (A-packed)
                          + 512 * 4                  // f2 row
                          + 512 * 4                  // mask row
                          + 64 * 4;                  // 1/sum per query row

// Fragment-packed layouts (wave32, ISA 7.12.2):
//  A-pack: chunk=(rowtile*KT+ktile); lane=((k>>3)&1)*16 + (row&15);
//          elem=(k&7) + ((k>>4)&1)*8
//  B-pack: chunk=(ktile*NT+ntile);  lane=((k>>3)&1)*16 + (col&15); elem=(k&7) + ((k>>4)&1)*8
// Each (chunk,lane) is 16 contiguous halfs (32 B) -> fragment load is 2x b128.

__device__ __forceinline__ v16h ldFrag(const _Float16* __restrict__ base, int chunk) {
  int lane = threadIdx.x & 31;
  return *(const v16h*)(base + (((size_t)chunk << 5) + lane) * 16);
}

// A fragment built from row-major f32 source (X stream), cvt on load.
__device__ __forceinline__ v16h ldA32(const float* __restrict__ A, int lda) {
  int lane = threadIdx.x & 31;
  const float* r = A + (size_t)(lane & 15) * lda + ((lane >> 4) << 3);
  v16h a;
#pragma unroll
  for (int v = 0; v < 8; ++v) {
    int k = (v < 4) ? (2 * v) : (8 + 2 * v);
    f2v t = *(const f2v*)(r + k);             // b64 load
    a[2 * v]     = (_Float16)t.x;
    a[2 * v + 1] = (_Float16)t.y;
  }
  return a;
}

__device__ __forceinline__ v8f wmma_f16(v16h a, v16h b, v8f c) {
  return __builtin_amdgcn_wmma_f32_16x16x32_f16(false, a, false, b,
                                                (short)0, c, false, false);
}

// ---------------- K0: pack f32 weights into B-fragment order ---------------
__global__ void __launch_bounds__(256)
k_packB(const float* __restrict__ src, _Float16* __restrict__ dst,
        int K, int Ncols, int chunksPerMat, int total) {
  int idx = blockIdx.x * blockDim.x + threadIdx.x;   // one 32B chunk each
  if (idx >= total) return;
  int mat = idx / chunksPerMat;
  int c   = idx % chunksPerMat;
  int lane = c & 31;
  int tile = c >> 5;
  int NT = Ncols >> 4;
  int nt = tile % NT, kt = tile / NT;
  int n = nt * 16 + (lane & 15);
  int kbase = kt * 32 + ((lane >> 4) & 1) * 8;
  const float* s = src + (size_t)mat * K * Ncols;
  v16h v;
#pragma unroll
  for (int e = 0; e < 16; ++e) v[e] = (_Float16)s[(size_t)(kbase + (e & 7) + ((e >> 3) << 4)) * Ncols + n];
  *(v16h*)(dst + (size_t)idx * 16) = v;
}

// Epilogue helper: store one 16x128 f32 accumulator strip into A-packed f16.
// rt = global row tile, KT = K/32 of the *next* GEMM that reads this buffer.
__device__ __forceinline__ void storeApk(_Float16* __restrict__ dstBase, int rt, int KT,
                                         const v8f* acc, const float* biasCol) {
  int lane = threadIdx.x & 31, n16 = lane & 15, hl = lane >> 4;
#pragma unroll
  for (int t = 0; t < 8; ++t) {
    int kt = t >> 1;
    int laneD = ((n16 >> 3) & 1) * 16;
    int elem = (n16 & 7) + (t & 1) * 8;
    float bb = biasCol ? biasCol[t * 16 + n16] : 0.f;
#pragma unroll
    for (int i = 0; i < 8; ++i) {
      int r15 = hl * 8 + i;
      size_t off = (((size_t)(rt * KT + kt) << 5) + laneD + r15) * 16 + elem;
      dstBase[off] = (_Float16)(acc[t][i] + bb);
    }
  }
}

// ---------------- K1: h = f16(X @ Wp + bp)   [16384,768]x[768,128] ---------
__global__ void __launch_bounds__(128)
k_proj(const float* __restrict__ X, const _Float16* __restrict__ Wp16,
       const float* __restrict__ bp, _Float16* __restrict__ hOut) {
  int wid = blockIdx.x * (blockDim.x >> 5) + (threadIdx.x >> 5);
  if (wid >= (B * N) / 16) return;            // 1024 row strips, wave-uniform
  int m0 = wid * 16;
  v8f acc[8] = {};
  for (int kt = 0; kt < D / 32; ++kt) {
    if (kt + 1 < D / 32)                      // prefetch next A k-slice (HBM stream)
      __builtin_prefetch(X + (size_t)(m0 + (threadIdx.x & 15)) * D + (kt + 1) * 32, 0, 0);
    v16h a = ldA32(X + (size_t)m0 * D + kt * 32, D);
#pragma unroll
    for (int t = 0; t < 8; ++t)
      acc[t] = wmma_f16(a, ldFrag(Wp16, kt * 8 + t), acc[t]);
  }
  storeApk(hOut, wid, R / 32, acc, bp);       // next GEMMs read h with K=128
}

// ---------------- K2: Wh = h @ gat_W[l,h]; fused f1/f2 dot products --------
__global__ void __launch_bounds__(128)
k_whf(const _Float16* __restrict__ hIn, const _Float16* __restrict__ gatW16,
      const float* __restrict__ ga1, const float* __restrict__ ga2,
      _Float16* __restrict__ WhOut, float* __restrict__ f1,
      float* __restrict__ f2, int layer) {
  int wid = blockIdx.x * (blockDim.x >> 5) + (threadIdx.x >> 5);
  if (wid >= B * H * (N / 16)) return;        // 4096 strips
  int mt = wid & (N / 16 - 1);                // 0..31
  int bh = wid >> 5;                          // 0..127
  int hh = bh & (H - 1);
  int b  = bh >> 2;
  const _Float16* W = gatW16 + (size_t)(layer * H + hh) * R * R;
  v8f acc[8] = {};
  for (int kt = 0; kt < R / 32; ++kt) {
    v16h a = ldFrag(hIn, (b * 32 + mt) * (R / 32) + kt);
#pragma unroll
    for (int t = 0; t < 8; ++t)
      acc[t] = wmma_f16(a, ldFrag(W, kt * 8 + t), acc[t]);
  }
  int lane = threadIdx.x & 31, n16 = lane & 15, hl = lane >> 4;
  const float* a1 = ga1 + (size_t)(layer * H + hh) * R;
  const float* a2 = ga2 + (size_t)(layer * H + hh) * R;

  // Store Wh in B-fragment order (it is the B operand of att @ Wh).
  // For row m = mt*16 + hl*8 + i: kt=m>>5 -> mt>>1; lane=hl*16+n16; elem=(mt&1)*8+i.
  _Float16* blk = WhOut + (size_t)bh * N * R;
  float p1[8] = {}, p2[8] = {};
#pragma unroll
  for (int t = 0; t < 8; ++t) {
    int col = t * 16 + n16;
    float w1 = a1[col], w2 = a2[col];
    v8h pk;
#pragma unroll
    for (int i = 0; i < 8; ++i) {
      float v = acc[t][i];
      pk[i] = (_Float16)v;
      p1[i] += v * w1;
      p2[i] += v * w2;
    }
    size_t off = (((size_t)((mt >> 1) * 8 + t) << 5) + hl * 16 + n16) * 16 + (mt & 1) * 8;
    *(v8h*)(blk + off) = pk;                  // 16 B vector store
  }
  size_t fBase = (size_t)bh * N + mt * 16;
#pragma unroll
  for (int i = 0; i < 8; ++i) {
    float s1 = p1[i], s2 = p2[i];
#pragma unroll
    for (int off = 1; off < 16; off <<= 1) {  // xor 1/2/4/8 keeps lane-half
      s1 += __shfl_xor(s1, off, 32);
      s2 += __shfl_xor(s2, off, 32);
    }
    if (n16 == 0) {
      f1[fBase + hl * 8 + i] = s1;
      f2[fBase + hl * 8 + i] = s2;
    }
  }
}

// ---------------- K3: fused softmax(leaky(f1+f2)) @ Wh, ELU, concat -------
// Wh tile (128 KB, already fragment-packed) staged via async-to-LDS DMA.
__global__ void __launch_bounds__(128)
k_attn(const _Float16* __restrict__ Wh, const float* __restrict__ f1,
       const float* __restrict__ f2, const float* __restrict__ mask,
       _Float16* __restrict__ cat) {
  extern __shared__ char smem[];
  _Float16* WhT  = (_Float16*)smem;                       // 512*128 halfs, packed
  _Float16* pBuf = (_Float16*)(smem + 131072);            // 4 waves x 8192 halfs
  float* f2s  = (float*)(smem + 131072 + 65536);
  float* msk  = f2s + 512;
  float* sInv = msk + 512;

  int tid = threadIdx.x;
  int chunk = blockIdx.x & 7;       // 8 chunks of 64 query rows
  int bh = blockIdx.x >> 3;
  int hh = bh & (H - 1);
  int b  = bh >> 2;

  // Linear async DMA of the packed Wh block: 8192 x 16 B, ASYNCcnt-tracked.
  {
    const char* gsrc = (const char*)(Wh + (size_t)bh * N * R);
    uint32_t ldsBase = (uint32_t)(uintptr_t)WhT;
    for (int idx = tid; idx < 512 * 16; idx += 128) {
      uint32_t loff = ldsBase + (uint32_t)(idx * 16);
      uint64_t ga = (uint64_t)(uintptr_t)(gsrc + (size_t)idx * 16);
      asm volatile("global_load_async_to_lds_b128 %0, %1, off"
                   :: "v"(loff), "v"(ga) : "memory");
    }
  }
  for (int j = tid; j < 512; j += 128) {
    f2s[j] = f2[(size_t)bh * N + j];
    msk[j] = mask[(size_t)b * N + j];
  }
  asm volatile("s_wait_asynccnt 0" ::: "memory");
  __syncthreads();

  int wave = tid >> 5, lane = tid & 31;
  int q0 = chunk * 64 + wave * 16;
  _Float16* prow0 = pBuf + (size_t)wave * 16 * 512;   // wave-private A-packed tile

  // Two-pass softmax per query row; p scattered into A-fragment order.
  for (int i = 0; i < 16; ++i) {
    int q = q0 + i;
    float f1v = f1[(size_t)bh * N + q];
    float mq  = mask[(size_t)b * N + q];
    float mx = -3.4e38f;
    for (int j = lane; j < 512; j += 32) {
      float v = f1v + f2s[j];
      v = v > 0.f ? v : ALPHA_SLOPE * v;
      float e = (mq * msk[j]) == 0.f ? NEGV : v;
      mx = fmaxf(mx, e);
    }
#pragma unroll
    for (int off = 1; off < 32; off <<= 1) mx = fmaxf(mx, __shfl_xor(mx, off, 32));
    float s = 0.f;
    for (int j = lane; j < 512; j += 32) {
      float v = f1v + f2s[j];
      v = v > 0.f ? v : ALPHA_SLOPE * v;
      float e = (mq * msk[j]) == 0.f ? NEGV : v;
      float p = __expf(e - mx);
      s += p;
      int kt = j >> 5;
      int laneD = ((j >> 3) & 1) * 16 + i;
      int elem = (j & 7) + ((j >> 4) & 1) * 8;
      prow0[((kt * 32 + laneD) << 4) + elem] = (_Float16)p;
    }
#pragma unroll
    for (int off = 1; off < 32; off <<= 1) s += __shfl_xor(s, off, 32);
    if (lane == 0) sInv[wave * 16 + i] = 1.f / s;
  }
  __syncthreads();

  // [16x512] @ [512x128] via WMMA; both operands packed in LDS.
  v8f acc[8] = {};
  for (int kt = 0; kt < 16; ++kt) {
    v16h a = ldFrag(prow0, kt);
#pragma unroll
    for (int t = 0; t < 8; ++t)
      acc[t] = wmma_f16(a, ldFrag(WhT, kt * 8 + t), acc[t]);
  }
  int n16 = lane & 15, hl = lane >> 4;
  // cat is the A operand of the head projection (K=HR=512): A-packed store.
  int rt = (b * N + q0) >> 4;                 // global row tile
#pragma unroll
  for (int t = 0; t < 8; ++t) {
    int colg = hh * R + t * 16 + n16;         // k-index in the next GEMM
    int kt = colg >> 5;
    int laneD = ((colg >> 3) & 1) * 16;
    int elem = (colg & 7) + ((colg >> 4) & 1) * 8;
#pragma unroll
    for (int i = 0; i < 8; ++i) {
      int lr = hl * 8 + i;
      float v = acc[t][i] * sInv[wave * 16 + lr];   // softmax normalization
      v = v > 0.f ? v : (__expf(v) - 1.f);          // ELU
      size_t off = (((size_t)(rt * 16 + kt) << 5) + laneD + lr) * 16 + elem;
      cat[off] = (_Float16)v;
    }
  }
}

// ---------------- K4: h = f16(cat @ head_W[l] + head_b[l]) ----------------
__global__ void __launch_bounds__(128)
k_headproj(const _Float16* __restrict__ cat, const _Float16* __restrict__ hW16,
           const float* __restrict__ hB, _Float16* __restrict__ hOut, int layer) {
  int wid = blockIdx.x * (blockDim.x >> 5) + (threadIdx.x >> 5);
  if (wid >= (B * N) / 16) return;
  const _Float16* Wm = hW16 + (size_t)layer * HR * R;
  const float* bb = hB + (size_t)layer * R;
  v8f acc[8] = {};
  for (int kt = 0; kt < HR / 32; ++kt) {
    v16h a = ldFrag(cat, wid * (HR / 32) + kt);
#pragma unroll
    for (int t = 0; t < 8; ++t)
      acc[t] = wmma_f16(a, ldFrag(Wm, kt * 8 + t), acc[t]);
  }
  storeApk(hOut, wid, R / 32, acc, bb);
}

// ---------------- K5: out = h @ Wout + bout + x ---------------------------
__global__ void __launch_bounds__(128)
k_out(const _Float16* __restrict__ hIn, const _Float16* __restrict__ Wout16,
      const float* __restrict__ bout, const float* __restrict__ X,
      float* __restrict__ out) {
  int wid = blockIdx.x * (blockDim.x >> 5) + (threadIdx.x >> 5);
  if (wid >= (B * N / 16) * (D / 128)) return;   // 1024 x 6 strips
  int cg = wid % (D / 128);
  int rt = wid / (D / 128);
  int m0 = rt * 16;
  v8f acc[8] = {};
  for (int kt = 0; kt < R / 32; ++kt) {
    v16h a = ldFrag(hIn, rt * (R / 32) + kt);
#pragma unroll
    for (int t = 0; t < 8; ++t)
      acc[t] = wmma_f16(a, ldFrag(Wout16, kt * (D / 16) + cg * 8 + t), acc[t]);
  }
  int lane = threadIdx.x & 31, n16 = lane & 15, hl = lane >> 4;
#pragma unroll
  for (int t = 0; t < 8; ++t) {
    int col = cg * 128 + t * 16 + n16;
    float bb = bout[col];
#pragma unroll
    for (int i = 0; i < 8; ++i) {
      size_t row = (size_t)(m0 + hl * 8 + i);
      out[row * D + col] = acc[t][i] + bb + X[row * D + col];
    }
  }
}

// ---------------- launcher ------------------------------------------------
extern "C" void kernel_launch(void* const* d_in, const int* in_sizes, int n_in,
                              void* d_out, int out_size, void* d_ws, size_t ws_size,
                              hipStream_t stream) {
  const float* X    = (const float*)d_in[0];
  const float* msk  = (const float*)d_in[1];
  const float* Wp   = (const float*)d_in[2];
  const float* bp   = (const float*)d_in[3];
  const float* gatW = (const float*)d_in[4];
  const float* ga1  = (const float*)d_in[5];
  const float* ga2  = (const float*)d_in[6];
  const float* hW   = (const float*)d_in[7];
  const float* hB   = (const float*)d_in[8];
  const float* Wout = (const float*)d_in[9];
  const float* boutp= (const float*)d_in[10];
  float* out = (float*)d_out;

  char* ws = (char*)d_ws;
  _Float16* hBuf = (_Float16*)ws;                        //  4 MB [16384,128] A-packed
  _Float16* catB = (_Float16*)(ws + ((size_t)4  << 20)); // 16 MB [16384,512] A-packed
  _Float16* WhB  = (_Float16*)(ws + ((size_t)20 << 20)); // 16 MB [128 blocks] B-packed
  float*    f1B  = (float*)   (ws + ((size_t)36 << 20)); // 256 KB
  float*    f2B  = (float*)   (ws + ((size_t)36 << 20) + (512 << 10));
  char*     w16  = ws + ((size_t)37 << 20);              // packed f16 weight pool
  _Float16* Wp16   = (_Float16*)w16;                       // 98304
  _Float16* gatW16 = (_Float16*)(w16 + 256 * 1024);        // 131072
  _Float16* hW16   = (_Float16*)(w16 + 640 * 1024);        // 131072
  _Float16* Wout16 = (_Float16*)(w16 + 1024 * 1024);       // 98304

  dim3 blk(128);
  // one-shot weight packing into B-fragment order
  k_packB<<<(D * R / 16 + 255) / 256, 256, 0, stream>>>(Wp, Wp16, D, R, D * R / 16, D * R / 16);
  k_packB<<<(NLAYER * H * R * R / 16 + 255) / 256, 256, 0, stream>>>(
      gatW, gatW16, R, R, R * R / 16, NLAYER * H * R * R / 16);
  k_packB<<<(NLAYER * HR * R / 16 + 255) / 256, 256, 0, stream>>>(
      hW, hW16, HR, R, HR * R / 16, NLAYER * HR * R / 16);
  k_packB<<<(R * D / 16 + 255) / 256, 256, 0, stream>>>(Wout, Wout16, R, D, R * D / 16, R * D / 16);

  // input projection: 1024 strips / 4 waves per block
  k_proj<<<256, blk, 0, stream>>>(X, Wp16, bp, hBuf);

  for (int l = 0; l < NLAYER; ++l) {
    // Wh + attention-logit halves: 4096 strips
    k_whf<<<1024, blk, 0, stream>>>(hBuf, gatW16, ga1, ga2, WhB, f1B, f2B, l);
    // fused attention: B*H*8 = 1024 workgroups, big-LDS + async-DMA path
    k_attn<<<1024, blk, ATTN_LDS, stream>>>(WhB, f1B, f2B, msk, catB);
    // head output projection
    k_headproj<<<256, blk, 0, stream>>>(catB, hW16, hB, hBuf, l);
  }

  // output projection + bias + residual: 6144 strips
  k_out<<<1536, blk, 0, stream>>>(hBuf, Wout16, boutp, X, out);
}
